// TransformerModel_30270929502938
// MI455X (gfx1250) — compile-verified, matrix operand order checked by reading, not run on hardware
//
#include <hip/hip_runtime.h>
#include <hip/hip_bf16.h>

typedef __attribute__((ext_vector_type(16))) _Float16 v16h;
typedef __attribute__((ext_vector_type(8)))  float    v8f;
typedef __attribute__((ext_vector_type(4)))  int      v4i_;

#define LNUM   6
#define FDIM   256
#define NHEAD  8
#define TDIM   256
#define CDIM   4
#define NBATCH 16
#define HDIM   32
#define EDIM   1024
#define BSEQ   64          // C*N sequences
#define MTOK   16384       // BSEQ * T tokens
#define DDIM   262144      // T*F*C
#define SPLITK 256

// ---------- optional CDNA5 async global->LDS path (guarded; safe fallback) ----------
#if defined(__HIP_DEVICE_COMPILE__) && defined(__gfx1250__)
#if __has_builtin(__builtin_amdgcn_global_load_async_to_lds_b128) && \
    __has_builtin(__builtin_amdgcn_s_wait_asynccnt)
#define HAS_ASYNC 1
#endif
#endif
#ifndef HAS_ASYNC
#define HAS_ASYNC 0
#endif

#if HAS_ASYNC
typedef __attribute__((address_space(1))) v4i_* gv4p;  // global 16B vector ptr
typedef __attribute__((address_space(3))) v4i_* lv4p;  // LDS 16B vector ptr
__device__ __forceinline__ void async_cp16(void* l, const void* g) {
  __builtin_amdgcn_global_load_async_to_lds_b128((gv4p)g, (lv4p)l, 0, 0);
}
__device__ __forceinline__ void async_wait0() { __builtin_amdgcn_s_wait_asynccnt(0); }
#endif

// ---- CDNA5 16-bit A/B fragment K-mapping (16x32 tile, wave32) ----
__device__ __forceinline__ int kmap_(int khalf, int hi) {
  int j = hi >> 1, p = hi & 1;
  int base = (j < 4) ? 0 : 16;
  return base + khalf * 8 + (j & 3) * 2 + p;
}

__device__ __forceinline__ v8f wmma16(v16h a, v16h b, v8f c) {
  return __builtin_amdgcn_wmma_f32_16x16x32_f16(false, a, false, b, (short)0, c,
                                                false, false);
}
__device__ __forceinline__ v8f v8f_zero() { v8f z = {}; return z; }

// pack 16 f32 (two contiguous 8-float runs at wp and wp+16) into one fragment chunk
__device__ __forceinline__ v16h pack_w16(const float* wp, bool ok) {
  float4 z4 = make_float4(0.f, 0.f, 0.f, 0.f);
  float4 w0 = ok ? *(const float4*)(wp + 0) : z4;
  float4 w1 = ok ? *(const float4*)(wp + 4) : z4;
  float4 w2 = ok ? *(const float4*)(wp + 16) : z4;
  float4 w3 = ok ? *(const float4*)(wp + 20) : z4;
  v16h wv;
  wv[0] = (_Float16)w0.x;  wv[1] = (_Float16)w0.y;
  wv[2] = (_Float16)w0.z;  wv[3] = (_Float16)w0.w;
  wv[4] = (_Float16)w1.x;  wv[5] = (_Float16)w1.y;
  wv[6] = (_Float16)w1.z;  wv[7] = (_Float16)w1.w;
  wv[8] = (_Float16)w2.x;  wv[9] = (_Float16)w2.y;
  wv[10] = (_Float16)w2.z; wv[11] = (_Float16)w2.w;
  wv[12] = (_Float16)w3.x; wv[13] = (_Float16)w3.y;
  wv[14] = (_Float16)w3.z; wv[15] = (_Float16)w3.w;
  return wv;
}

// ============================ prep: x -> h (+pos), f32+f16 ============================
__global__ __launch_bounds__(256) void prep_kernel(const float* __restrict__ x,
                                                   float* __restrict__ h,
                                                   _Float16* __restrict__ hh) {
  int idx = blockIdx.x * 256 + threadIdx.x;      // 64*256*256 total
  int f = idx & 255;
  int t = (idx >> 8) & 255;
  int row = idx >> 16;                           // c*16 + n
  int c = row >> 4, n = row & 15;
  float val = x[(((size_t)n * FDIM + f) * TDIM + t) * CDIM + c] + (float)t * (1.0f / 255.0f);
  h[idx] = val;
  hh[idx] = (_Float16)val;
}

// ============================ generic GEMM: Y = act(alpha*X*W^T + b [+ res]) =========
// X: (M,K) f16 row-major; W: (N,K) f32 row-major; 4 waves, 64x64 tile, K step 32.
template <int TYF16, int ACTRELU>
__global__ __launch_bounds__(128) void gemm_xwT(const _Float16* __restrict__ X,
                                                const float* __restrict__ W,
                                                const float* __restrict__ bias,
                                                const float* __restrict__ resid,
                                                void* __restrict__ Y, int M, int Nn,
                                                int K, float alpha) {
  __shared__ __align__(32) _Float16 Xs[2048];  // 4 frag-tiles of 16x32
  __shared__ __align__(32) _Float16 Ws[2048];
  int tid = threadIdx.x;
  int w = tid >> 5, lane = tid & 31;
  int ft = tid >> 5, rr = tid & 31;
  int khalf = rr >> 4, srow = rr & 15;
  int my0 = blockIdx.y * 64, nx0 = blockIdx.x * 64;

  v8f acc[4];
#pragma unroll
  for (int i = 0; i < 4; ++i) acc[i] = v8f_zero();

  for (int k0 = 0; k0 < K; k0 += 32) {
    int xr = my0 + ft * 16 + srow;
    int wr = nx0 + ft * 16 + srow;
    bool xok = (xr < M), wok = (wr < Nn);
    // ---- X tile: two contiguous 16B chunks per lane into fragment order
    const _Float16* xp = X + (size_t)xr * K + k0 + khalf * 8;
#if HAS_ASYNC
    if (xok) {
      async_cp16(Xs + tid * 16, xp);
      async_cp16(Xs + tid * 16 + 8, xp + 16);
    } else {
      uint4 z = make_uint4(0u, 0u, 0u, 0u);
      *(uint4*)(Xs + tid * 16) = z;
      *(uint4*)(Xs + tid * 16 + 8) = z;
    }
#else
    {
      uint4 z = make_uint4(0u, 0u, 0u, 0u);
      uint4 a0 = xok ? *(const uint4*)(xp) : z;
      uint4 a1 = xok ? *(const uint4*)(xp + 16) : z;
      *(uint4*)(Xs + tid * 16) = a0;
      *(uint4*)(Xs + tid * 16 + 8) = a1;
    }
#endif
    // ---- W tile: 4 float4 loads, convert, one 32B store
    {
      const float* wp = W + (size_t)wr * K + k0 + khalf * 8;
      if (wok) __builtin_prefetch(wp + 32, 0, 0);
      *(v16h*)(Ws + tid * 16) = pack_w16(wp, wok);
    }
#if HAS_ASYNC
    async_wait0();
#endif
    __syncthreads();
    v16h a = *(const v16h*)(Xs + w * 512 + lane * 16);
#pragma unroll
    for (int nt = 0; nt < 4; ++nt) {
      v16h b = *(const v16h*)(Ws + nt * 512 + lane * 16);
      acc[nt] = wmma16(a, b, acc[nt]);
    }
    __syncthreads();
  }

  // Epilogue (C/D layout: vgpr r -> M = r + 8*(lane>=16); N = lane&15)
  int rowoff = (lane >> 4) << 3;
  int l15 = lane & 15;
#pragma unroll
  for (int nt = 0; nt < 4; ++nt) {
    int col = nx0 + nt * 16 + l15;
    if (col >= Nn) continue;
    float bval = bias ? bias[col] : 0.0f;
#pragma unroll
    for (int r = 0; r < 8; ++r) {
      int m = my0 + w * 16 + rowoff + r;
      if (m >= M) continue;
      float vv = acc[nt][r] * alpha + bval;
      if (resid) vv += resid[(size_t)m * Nn + col];
      if (ACTRELU) vv = fmaxf(vv, 0.0f);
      if (TYF16)
        ((_Float16*)Y)[(size_t)m * Nn + col] = (_Float16)vv;
      else
        ((float*)Y)[(size_t)m * Nn + col] = vv;
    }
  }
}

// ============================ attention, one (seq,head) per block ====================
// q pre-scaled by 1/sqrt(F). Resident two-pass softmax (T=256 keys in registers).
#define AQ 0
#define AK 8192
#define AV 16384
#define AP 24576
__global__ __launch_bounds__(256) void attn_kernel(const _Float16* __restrict__ q,
                                                   const _Float16* __restrict__ k,
                                                   const _Float16* __restrict__ v,
                                                   _Float16* __restrict__ o) {
  __shared__ __align__(32) _Float16 lds[24576 + 8 * 512];  // 57KB
  int n = blockIdx.x >> 3, hd = blockIdx.x & 7;
  const size_t base = ((size_t)n * TDIM) * FDIM + hd * HDIM;

  // Q,K: contiguous 16B chunks into fragment order
  for (int u = threadIdx.x; u < 512; u += 256) {
    int tile = u >> 5, kh = (u >> 4) & 1, row = u & 15;
    const size_t so = base + (size_t)(tile * 16 + row) * FDIM + kh * 8;
#if HAS_ASYNC
    async_cp16(lds + AQ + u * 16, q + so);
    async_cp16(lds + AQ + u * 16 + 8, q + so + 16);
    async_cp16(lds + AK + u * 16, k + so);
    async_cp16(lds + AK + u * 16 + 8, k + so + 16);
#else
    *(uint4*)(lds + AQ + u * 16) = *(const uint4*)(q + so);
    *(uint4*)(lds + AQ + u * 16 + 8) = *(const uint4*)(q + so + 16);
    *(uint4*)(lds + AK + u * 16) = *(const uint4*)(k + so);
    *(uint4*)(lds + AK + u * 16 + 8) = *(const uint4*)(k + so + 16);
#endif
  }
  // V: transpose gather (B-fragment wants V^T)
  for (int idx = threadIdx.x; idx < 8192; idx += 256) {
    int tile = idx >> 9, rem = idx & 511;
    int khr = rem >> 4, hi = rem & 15;
    int kh = khr >> 4, row = khr & 15;
    int km = kmap_(kh, hi);
    int kb5 = tile >> 1, dt = tile & 1;
    lds[AV + idx] = v[base + (size_t)(kb5 * 32 + km) * FDIM + dt * 16 + row];
  }
#if HAS_ASYNC
  async_wait0();
#endif
  __syncthreads();

  int w = threadIdx.x >> 5, lane = threadIdx.x & 31;
  int l15 = lane & 15, rowoff = (lane >> 4) << 3;
  int kh0 = l15 >> 3, hi0 = l15 & 7;  // P scatter map (col<16); +16 cols use hi0+8
  _Float16* Pb = lds + AP + w * 512;

  for (int qt = w; qt < 16; qt += 8) {
    v16h qa = *(const v16h*)(lds + AQ + qt * 512 + lane * 16);
    v8f S[16];
#pragma unroll
    for (int jt = 0; jt < 16; ++jt) {
      v16h b = *(const v16h*)(lds + AK + jt * 512 + lane * 16);
      S[jt] = wmma16(qa, b, v8f_zero());
    }
    float mx[8], sm[8];
#pragma unroll
    for (int r = 0; r < 8; ++r) {
      float t = S[0][r];
#pragma unroll
      for (int jt = 1; jt < 16; ++jt) t = fmaxf(t, S[jt][r]);
      t = fmaxf(t, __shfl_xor(t, 1, 16));
      t = fmaxf(t, __shfl_xor(t, 2, 16));
      t = fmaxf(t, __shfl_xor(t, 4, 16));
      t = fmaxf(t, __shfl_xor(t, 8, 16));
      mx[r] = t;
    }
#pragma unroll
    for (int r = 0; r < 8; ++r) {
      float t = 0.0f;
#pragma unroll
      for (int jt = 0; jt < 16; ++jt) {
        S[jt][r] = __expf(S[jt][r] - mx[r]);
        t += S[jt][r];
      }
      t += __shfl_xor(t, 1, 16);
      t += __shfl_xor(t, 2, 16);
      t += __shfl_xor(t, 4, 16);
      t += __shfl_xor(t, 8, 16);
      sm[r] = t;
    }
    v8f O0 = v8f_zero(), O1 = v8f_zero();
#pragma unroll
    for (int kb = 0; kb < 8; ++kb) {
#pragma unroll
      for (int r = 0; r < 8; ++r) {
        int prow = rowoff + r;
        Pb[(kh0 * 16 + prow) * 16 + hi0] = (_Float16)S[2 * kb][r];
        Pb[(kh0 * 16 + prow) * 16 + hi0 + 8] = (_Float16)S[2 * kb + 1][r];
      }
      v16h pa = *(const v16h*)(Pb + lane * 16);
      v16h vb0 = *(const v16h*)(lds + AV + (kb * 2 + 0) * 512 + lane * 16);
      v16h vb1 = *(const v16h*)(lds + AV + (kb * 2 + 1) * 512 + lane * 16);
      O0 = wmma16(pa, vb0, O0);
      O1 = wmma16(pa, vb1, O1);
    }
#pragma unroll
    for (int r = 0; r < 8; ++r) {
      float inv = 1.0f / sm[r];
      int t = qt * 16 + rowoff + r;
      o[base + (size_t)t * FDIM + l15] = (_Float16)(O0[r] * inv);
      o[base + (size_t)t * FDIM + 16 + l15] = (_Float16)(O1[r] * inv);
    }
  }
}

// ============================ LayerNorm over F=256, writes f32 + f16 =================
__global__ __launch_bounds__(256) void ln_kernel(const float* __restrict__ in,
                                                 const float* __restrict__ g,
                                                 const float* __restrict__ b,
                                                 float* __restrict__ of32,
                                                 _Float16* __restrict__ of16) {
  __shared__ float rs[8], rq[8];
  int row = blockIdx.x, tid = threadIdx.x;
  int w = tid >> 5, lane = tid & 31;
  float v = in[(size_t)row * FDIM + tid];
  float s = v, q = v * v;
#pragma unroll
  for (int off = 16; off >= 1; off >>= 1) {
    s += __shfl_xor(s, off, 32);
    q += __shfl_xor(q, off, 32);
  }
  if (lane == 0) { rs[w] = s; rq[w] = q; }
  __syncthreads();
  float S = 0.0f, Q = 0.0f;
#pragma unroll
  for (int i = 0; i < 8; ++i) { S += rs[i]; Q += rq[i]; }
  float m = S * (1.0f / FDIM);
  float var = Q * (1.0f / FDIM) - m * m;
  float rstd = rsqrtf(var + 1e-5f);
  float y = (v - m) * rstd * g[tid] + b[tid];
  of32[(size_t)row * FDIM + tid] = y;
  of16[(size_t)row * FDIM + tid] = (_Float16)y;
}

// ============================ permute encoder output to flat (f16) ===================
__global__ __launch_bounds__(256) void permute_kernel(const float* __restrict__ h,
                                                      _Float16* __restrict__ flat) {
  int idx = blockIdx.x * 256 + threadIdx.x;  // 16*262144
  int n = idx >> 18;
  int rem = idx & (DDIM - 1);
  int t = rem >> 10, fc = rem & 1023;
  int f = fc >> 2, c = fc & 3;
  flat[idx] = (_Float16)h[((size_t)(c * NBATCH + n) * TDIM + t) * FDIM + f];
}

// ============================ head stem: split-K WMMA (M=16) =========================
__global__ __launch_bounds__(128) void gemm_head_splitk(const _Float16* __restrict__ X,
                                                        const float* __restrict__ W,
                                                        float* __restrict__ part, int K) {
  __shared__ __align__(32) _Float16 Xs[512];
  __shared__ __align__(32) _Float16 Ws[2048];
  int tid = threadIdx.x;
  int w = tid >> 5, lane = tid & 31;
  int rr = tid & 31;
  int khalf = rr >> 4, srow = rr & 15;
  int nx0 = blockIdx.x * 64;
  int kc = K / SPLITK;  // 1024
  int kbeg = blockIdx.y * kc;
  v8f acc = v8f_zero();

  for (int ks = 0; ks < kc; ks += 32) {
    int k0 = kbeg + ks;
    if (tid < 32) {
      const _Float16* xp = X + (size_t)srow * K + k0 + khalf * 8;
#if HAS_ASYNC
      async_cp16(Xs + tid * 16, xp);
      async_cp16(Xs + tid * 16 + 8, xp + 16);
#else
      *(uint4*)(Xs + tid * 16) = *(const uint4*)(xp);
      *(uint4*)(Xs + tid * 16 + 8) = *(const uint4*)(xp + 16);
#endif
    }
    {
      int wr = nx0 + (tid >> 5) * 16 + srow;
      const float* wp = W + (size_t)wr * K + k0 + khalf * 8;
      __builtin_prefetch(wp + 32, 0, 0);
      *(v16h*)(Ws + tid * 16) = pack_w16(wp, true);
    }
#if HAS_ASYNC
    async_wait0();
#endif
    __syncthreads();
    v16h a = *(const v16h*)(Xs + lane * 16);
    v16h b = *(const v16h*)(Ws + w * 512 + lane * 16);
    acc = wmma16(a, b, acc);
    __syncthreads();
  }
  int rowoff = (lane >> 4) << 3, l15 = lane & 15;
#pragma unroll
  for (int r = 0; r < 8; ++r) {
    int m = rowoff + r;  // batch row (<16)
    int col = nx0 + w * 16 + l15;
    part[((size_t)blockIdx.y * 16 + m) * 256 + col] = acc[r];
  }
}

// reduce split-K partials + bias + eval-BN + relu -> f16
__global__ __launch_bounds__(256) void reduce_bn_kernel(const float* __restrict__ part,
                                                        const float* __restrict__ b1,
                                                        const float* __restrict__ g,
                                                        const float* __restrict__ bt,
                                                        _Float16* __restrict__ z) {
  int n = blockIdx.x, col = threadIdx.x;
  float s = 0.0f;
  for (int sk = 0; sk < SPLITK; ++sk) s += part[((size_t)sk * 16 + n) * 256 + col];
  s += b1[col];
  s = s * rsqrtf(1.0f + 1e-5f) * g[col] + bt[col];
  s = fmaxf(s, 0.0f);
  z[n * 256 + col] = (_Float16)s;
}

// clip + interleave heads into (N, 4)
__global__ __launch_bounds__(64) void finalize_kernel(const float* __restrict__ ze,
                                                      const float* __restrict__ za,
                                                      float* __restrict__ out) {
  const float PI = 3.14159265358979323846f;
  int tid = threadIdx.x;  // 64 = 16*4
  int n = tid >> 2, cc = tid & 3;
  int i = cc >> 1, az = cc & 1;
  float v;
  if (az) v = fminf(fmaxf(za[n * 2 + i], 0.0f), 2.0f * PI);
  else    v = fminf(fmaxf(ze[n * 2 + i], -PI * 0.25f), PI * 0.5f);
  out[tid] = v;
}

// =====================================================================================
extern "C" void kernel_launch(void* const* d_in, const int* in_sizes, int n_in,
                              void* d_out, int out_size, void* d_ws, size_t ws_size,
                              hipStream_t stream) {
  const float* x    = (const float*)d_in[0];
  const float* Wq   = (const float*)d_in[1];
  const float* Wk   = (const float*)d_in[2];
  const float* Wv   = (const float*)d_in[3];
  const float* Wo   = (const float*)d_in[4];
  const float* bo   = (const float*)d_in[5];
  const float* ln1g = (const float*)d_in[6];
  const float* ln1b = (const float*)d_in[7];
  const float* ln2g = (const float*)d_in[8];
  const float* ln2b = (const float*)d_in[9];
  const float* fW1  = (const float*)d_in[10];
  const float* fb1  = (const float*)d_in[11];
  const float* fW2  = (const float*)d_in[12];
  const float* fb2  = (const float*)d_in[13];
  const float* hw[2][9];  // [head][W1,b1,g,bt,W2,b2,W3,b3,W4]
  for (int hdx = 0; hdx < 2; ++hdx)
    for (int j = 0; j < 9; ++j) hw[hdx][j] = (const float*)d_in[14 + hdx * 9 + j];

  char* ws = (char*)d_ws;
  float*    h    = (float*)(ws + 0);                       // 16 MB
  _Float16* hh   = (_Float16*)(ws + 16777216);             // 8 MB
  _Float16* qb   = (_Float16*)(ws + 25165824);             // 8 MB (later: flat)
  _Float16* kb   = (_Float16*)(ws + 33554432);             // 8 MB (later: partials)
  _Float16* vb   = (_Float16*)(ws + 41943040);             // 8 MB (later: head temps)
  _Float16* ob   = (_Float16*)(ws + 50331648);             // 8 MB
  float*    t0   = (float*)(ws + 58720256);                // 16 MB
  float*    x1   = (float*)(ws + 75497472);                // 16 MB
  _Float16* x1h  = (_Float16*)(ws + 92274688);             // 8 MB
  _Float16* ffh  = (_Float16*)(ws + 100663296);            // 32 MB -> end 134217728
  _Float16* flat = qb;
  float*    part = (float*)kb;                             // 4 MB
  _Float16* z1   = vb;
  _Float16* z2   = (_Float16*)(ws + 41943040 + 8192);
  _Float16* z3   = (_Float16*)(ws + 41943040 + 16384);
  float*    z4e  = (float*)(ws + 41943040 + 24576);
  float*    z4a  = (float*)(ws + 41943040 + 24832);

  const float qscale = 1.0f / 16.0f;  // 1/sqrt(F)

  prep_kernel<<<MTOK, 256, 0, stream>>>(x, h, hh);

  for (int l = 0; l < LNUM; ++l) {
    // QKV projections: X = hh viewed as (MTOK*8, 32); q pre-scaled.
    dim3 gq(1, (MTOK * 8) / 64);
    gemm_xwT<1, 0><<<gq, 128, 0, stream>>>(hh, Wq + l * 1024, nullptr, nullptr, qb,
                                           MTOK * 8, 32, 32, qscale);
    gemm_xwT<1, 0><<<gq, 128, 0, stream>>>(hh, Wk + l * 1024, nullptr, nullptr, kb,
                                           MTOK * 8, 32, 32, 1.0f);
    gemm_xwT<1, 0><<<gq, 128, 0, stream>>>(hh, Wv + l * 1024, nullptr, nullptr, vb,
                                           MTOK * 8, 32, 32, 1.0f);
    attn_kernel<<<BSEQ * NHEAD, 256, 0, stream>>>(qb, kb, vb, ob);
    // Wo projection + residual
    gemm_xwT<0, 0><<<dim3(4, MTOK / 64), 128, 0, stream>>>(
        ob, Wo + (size_t)l * 65536, bo + l * 256, h, t0, MTOK, 256, 256, 1.0f);
    ln_kernel<<<MTOK, 256, 0, stream>>>(t0, ln1g + l * 256, ln1b + l * 256, x1, x1h);
    // FFN
    gemm_xwT<1, 1><<<dim3(16, MTOK / 64), 128, 0, stream>>>(
        x1h, fW1 + (size_t)l * EDIM * 256, fb1 + l * EDIM, nullptr, ffh, MTOK, EDIM,
        256, 1.0f);
    gemm_xwT<0, 0><<<dim3(4, MTOK / 64), 128, 0, stream>>>(
        ffh, fW2 + (size_t)l * 256 * EDIM, fb2 + l * 256, x1, t0, MTOK, 256, EDIM,
        1.0f);
    ln_kernel<<<MTOK, 256, 0, stream>>>(t0, ln2g + l * 256, ln2b + l * 256, h, hh);
  }

  permute_kernel<<<(NBATCH * DDIM) / 256, 256, 0, stream>>>(h, flat);

  for (int hdx = 0; hdx < 2; ++hdx) {
    float* z4 = hdx ? z4a : z4e;
    gemm_head_splitk<<<dim3(4, SPLITK), 128, 0, stream>>>(flat, hw[hdx][0], part, DDIM);
    reduce_bn_kernel<<<NBATCH, 256, 0, stream>>>(part, hw[hdx][1], hw[hdx][2],
                                                 hw[hdx][3], z1);
    gemm_xwT<1, 1><<<dim3(4, 1), 128, 0, stream>>>(z1, hw[hdx][4], hw[hdx][5], nullptr,
                                                   z2, NBATCH, 256, 256, 1.0f);
    gemm_xwT<1, 1><<<dim3(4, 1), 128, 0, stream>>>(z2, hw[hdx][6], hw[hdx][7], nullptr,
                                                   z3, NBATCH, 256, 256, 1.0f);
    gemm_xwT<0, 0><<<dim3(1, 1), 128, 0, stream>>>(z3, hw[hdx][8], nullptr, nullptr,
                                                   z4, NBATCH, 2, 256, 1.0f);
  }
  finalize_kernel<<<1, 64, 0, stream>>>(z4e, z4a, (float*)d_out);
}
